// ROIHead_81174881894443
// MI455X (gfx1250) — matrix operand order repaired and checked by reading, not compiled, not run on hardware
//
#include <hip/hip_runtime.h>
#include <hip/hip_bf16.h>

// ---------------- problem constants (from reference) ----------------
constexpr int BB   = 4;
constexpr int NN   = 2000;
constexpr int GG   = 20;
constexpr int ROWS = BB * NN;        // 8000 proposals
constexpr int FD   = 256 * 7 * 7;    // 12544 feature dim
constexpr int HID  = 1024;
constexpr int NCLS = 81;             // NUM_LABELS + 1
constexpr int HEADN = NCLS + 4;      // 85 (scores ++ deltas)
constexpr int NUM_POS = 128;         // 512 * 0.25
constexpr int NUM_NEG = 384;
constexpr float THR  = 0.5f;
constexpr float SCALE = 1.0f / (BB * 512.0f);   // 1/2048

// ---------------- vector types ----------------
typedef __attribute__((ext_vector_type(16))) __bf16 v16bf;
typedef __attribute__((ext_vector_type(8)))  __bf16 v8bf;
typedef __attribute__((ext_vector_type(4)))  __bf16 v4bf;
typedef __attribute__((ext_vector_type(8)))  float  v8f;

// gfx1250 async global->LDS path (ASYNCcnt), with synchronous fallback.
// Builtin signature (from hipcc diagnostics): (int4 AS1*, int4 AS3*, imm, imm)
#if __has_builtin(__builtin_amdgcn_global_load_async_to_lds_b128) && \
    __has_builtin(__builtin_amdgcn_s_wait_asynccnt)
#define HAVE_ASYNC_LDS 1
typedef int v4i_g __attribute__((__vector_size__(16)));
typedef __attribute__((address_space(1))) v4i_g* gbl_i4p;
typedef __attribute__((address_space(3))) v4i_g* lds_i4p;
#else
#define HAVE_ASYNC_LDS 0
#endif

// ---------------- conversion / packing kernels ----------------
__global__ void cvt_f32_bf16_vec4(const float* __restrict__ in,
                                  __bf16* __restrict__ out, long n4) {
  long t = (long)blockIdx.x * blockDim.x + threadIdx.x;
  if (t >= n4) return;
  float4 v = ((const float4*)in)[t];
  v4bf r;
  r[0] = (__bf16)v.x; r[1] = (__bf16)v.y; r[2] = (__bf16)v.z; r[3] = (__bf16)v.w;
  ((v4bf*)out)[t] = r;
}

// W[K,N] fp32 -> Wt[N,K] bf16, writes coalesced (k fastest)
__global__ void transpose_to_bf16(const float* __restrict__ in,
                                  __bf16* __restrict__ out,
                                  int K, int N, long total) {
  long t = (long)blockIdx.x * blockDim.x + threadIdx.x;
  if (t >= total) return;
  int n = (int)(t / K);
  int k = (int)(t % K);
  out[(size_t)n * K + k] = (__bf16)in[(size_t)k * N + n];
}

// pack [Wc | Wb] -> WhT[85,1024] fp32 + bias[85]
__global__ void pack_head(const float* __restrict__ Wc, const float* __restrict__ bc,
                          const float* __restrict__ Wb, const float* __restrict__ bbv,
                          float* __restrict__ WhT, float* __restrict__ bh) {
  int t = blockIdx.x * blockDim.x + threadIdx.x;
  if (t >= HEADN * HID) return;
  int col = t / HID, k = t % HID;
  WhT[(size_t)col * HID + k] = (col < NCLS) ? Wc[(size_t)k * NCLS + col]
                                            : Wb[(size_t)k * 4 + (col - NCLS)];
  if (k == 0) bh[col] = (col < NCLS) ? bc[col] : bbv[col - NCLS];
}

// ---------------- WMMA fragment loads (work on global or LDS pointers) ----------------
__device__ inline v16bf load_a_frag(const __bf16* p) {  // 16 contiguous along K
  v8bf lo = *(const v8bf*)p;
  v8bf hi = *(const v8bf*)(p + 8);
  v16bf r;
#pragma unroll
  for (int i = 0; i < 8; ++i) { r[i] = lo[i]; r[i + 8] = hi[i]; }
  return r;
}
__device__ inline v16bf load_b_frag(const __bf16* p) {  // 16 contiguous along K
  v8bf lo = *(const v8bf*)p;
  v8bf hi = *(const v8bf*)(p + 8);
  v16bf r;
#pragma unroll
  for (int i = 0; i < 8; ++i) { r[i] = lo[i]; r[i + 8] = hi[i]; }
  return r;
}

// ---------------- WMMA bf16 GEMM: C = relu(A[M,K] * Bt[N,K]^T + bias) ----------------
// Block: 256 threads (8 waves). Tile: M=128, N=64, K-step 32.
// A/B tiles double-buffered in LDS, filled by async global->LDS copies.
__global__ void __launch_bounds__(256)
gemm_bf16_relu(const __bf16* __restrict__ A,    // [M,K] row-major bf16
               const __bf16* __restrict__ Bt,   // [N,K] row-major bf16 (W transposed)
               const float*  __restrict__ bias, // [N]
               float*  __restrict__ Cf,         // fp32 out (or null)
               __bf16* __restrict__ Cb,         // bf16 out (or null)
               int M, int N, int K) {
  __shared__ __align__(16) __bf16 aT[2][128 * 32];   // 8 KB per buffer
  __shared__ __align__(16) __bf16 bT[2][64 * 32];    // 4 KB per buffer

  const int tid  = threadIdx.x;
  const int wave = tid >> 5;
  const int lane = tid & 31;
  const int rowBase = blockIdx.y * 128;
  const int colBase = blockIdx.x * 64;
  const int rowTile = rowBase + wave * 16;

  // 16-byte copy chunks: A tile = 512 chunks (2/thread), B tile = 256 (1/thread)
  const int arow0 = tid >> 2;                  // 0..63
  const int arow1 = (tid + 256) >> 2;          // 64..127
  const int apart = (tid & 3) << 3;            // element offset 0/8/16/24
  const int bcol  = tid >> 2;                  // 0..63
  const int bpart = (tid & 3) << 3;
  const int aSrc0 = (rowBase + arow0 < M) ? (rowBase + arow0) : (M - 1);
  const int aSrc1 = (rowBase + arow1 < M) ? (rowBase + arow1) : (M - 1);

  auto stage = [&](int k0, int buf) {
#if HAVE_ASYNC_LDS
    __builtin_amdgcn_global_load_async_to_lds_b128(
        (gbl_i4p)(A + (size_t)aSrc0 * K + k0 + apart),
        (lds_i4p)(&aT[buf][arow0 * 32 + apart]), 0, 0);
    __builtin_amdgcn_global_load_async_to_lds_b128(
        (gbl_i4p)(A + (size_t)aSrc1 * K + k0 + apart),
        (lds_i4p)(&aT[buf][arow1 * 32 + apart]), 0, 0);
    __builtin_amdgcn_global_load_async_to_lds_b128(
        (gbl_i4p)(Bt + (size_t)(colBase + bcol) * K + k0 + bpart),
        (lds_i4p)(&bT[buf][bcol * 32 + bpart]), 0, 0);
#else
    *(v8bf*)&aT[buf][arow0 * 32 + apart] =
        *(const v8bf*)(A + (size_t)aSrc0 * K + k0 + apart);
    *(v8bf*)&aT[buf][arow1 * 32 + apart] =
        *(const v8bf*)(A + (size_t)aSrc1 * K + k0 + apart);
    *(v8bf*)&bT[buf][bcol * 32 + bpart] =
        *(const v8bf*)(Bt + (size_t)(colBase + bcol) * K + k0 + bpart);
#endif
  };

  // fragment offsets inside the LDS tiles
  const int fRow  = wave * 16 + (lane & 15);
  const int aOff  = fRow * 32 + ((lane >> 4) << 4);       // row*32 + (0|16)
  const int bLane = (lane & 15) * 32 + ((lane >> 4) << 4);// col*32 + (0|16)

  v8f acc[4];
#pragma unroll
  for (int c = 0; c < 4; ++c) acc[c] = (v8f){};

  const int nk = K >> 5;
  stage(0, 0);
  for (int i = 0; i < nk; ++i) {
    const int cur = i & 1;
    if (i + 1 < nk) {
      stage((i + 1) << 5, cur ^ 1);     // async copies for next stage in flight
#if HAVE_ASYNC_LDS
      __builtin_amdgcn_s_wait_asynccnt(3);   // current stage's 3 copies done
#endif
    } else {
#if HAVE_ASYNC_LDS
      __builtin_amdgcn_s_wait_asynccnt(0);
#endif
    }
    __syncthreads();                     // tile visible to all waves
    v16bf a = load_a_frag(&aT[cur][aOff]);
#pragma unroll
    for (int c = 0; c < 4; ++c) {
      v16bf b = load_b_frag(&bT[cur][c * 16 * 32 + bLane]);
      acc[c] = __builtin_amdgcn_wmma_f32_16x16x32_bf16(
          false, a, false, b, (short)0, acc[c], false, false);
    }
    __syncthreads();                     // done reading before buffer reuse
  }

  if (rowTile < M) {                     // uniform per wave
    const int rOff  = (lane >> 4) << 3;  // C/D layout: lanes 16-31 hold M+8
    const int cLane = lane & 15;
#pragma unroll
    for (int c = 0; c < 4; ++c) {
#pragma unroll
      for (int r = 0; r < 8; ++r) {
        int row = rowTile + rOff + r;
        int col = colBase + c * 16 + cLane;
        float v = acc[c][r] + bias[col];
        v = fmaxf(v, 0.0f);
        if (Cf) Cf[(size_t)row * N + col] = v;
        if (Cb) Cb[(size_t)row * N + col] = (__bf16)v;
      }
    }
  }
}

// ---------------- small head: logits[row,85] = h2[row,:] . WhT[col,:] + bh ----------------
__global__ void head_kernel(const float* __restrict__ h2, const float* __restrict__ WhT,
                            const float* __restrict__ bh, float* __restrict__ logits) {
  long t = (long)blockIdx.x * blockDim.x + threadIdx.x;
  if (t >= (long)ROWS * HEADN) return;
  int row = (int)(t / HEADN), col = (int)(t % HEADN);
  const float* hp = h2 + (size_t)row * HID;
  const float* wp = WhT + (size_t)col * HID;
  float s = bh[col];
#pragma unroll 4
  for (int k = 0; k < HID; ++k) s += hp[k] * wp[k];
  logits[(size_t)row * HEADN + col] = s;
}

// ---------------- labeling: IoU argmax + threshold ----------------
__global__ void label_kernel(const float* __restrict__ proposals,
                             const float* __restrict__ gt_boxes,
                             const int*   __restrict__ gt_labels,
                             int* __restrict__ label, float* __restrict__ mgt) {
  int t = blockIdx.x * blockDim.x + threadIdx.x;
  if (t >= ROWS) return;
  int b = t / NN;
  const float* p = proposals + (size_t)t * 4;
  float p0 = p[0], p1 = p[1], p2 = p[2], p3 = p[3];
  float ap = (p2 - p0) * (p3 - p1);
  float best = -1.0f; int bi = 0;
  for (int g = 0; g < GG; ++g) {
    const float* q = gt_boxes + ((size_t)b * GG + g) * 4;
    float x1 = fmaxf(p0, q[0]), y1 = fmaxf(p1, q[1]);
    float x2 = fminf(p2, q[2]), y2 = fminf(p3, q[3]);
    float inter = fmaxf(x2 - x1, 0.0f) * fmaxf(y2 - y1, 0.0f);
    float ag = (q[2] - q[0]) * (q[3] - q[1]);
    float iou = inter / (ap + ag - inter);
    if (iou > best) { best = iou; bi = g; }        // first-max, like argmax
  }
  bool fg = best > THR;
  label[t] = fg ? gt_labels[(size_t)b * GG + bi] : 0;
  const float* q = gt_boxes + ((size_t)b * GG + bi) * 4;
  float* m = mgt + (size_t)t * 4;
  m[0] = fg ? q[0] : 0.0f; m[1] = fg ? q[1] : 0.0f;
  m[2] = fg ? q[2] : 0.0f; m[3] = fg ? q[3] : 0.0f;
}

// ---------------- sequential quota sampling (cumsum semantics) ----------------
__global__ void sample_kernel(const int* __restrict__ label, int* __restrict__ samp) {
  int b = blockIdx.x * blockDim.x + threadIdx.x;
  if (b >= BB) return;
  int cp = 0, cn = 0;
  for (int n = 0; n < NN; ++n) {
    int l = label[b * NN + n], s;
    if (l >= 1) { ++cp; s = (cp <= NUM_POS) ? l : -1; }
    else        { ++cn; s = (cn <= NUM_NEG) ? 0 : -1; }
    samp[b * NN + n] = s;
  }
}

// ---------------- fused loss (double softmax + smooth-L1), per-block partials ----------------
__global__ void __launch_bounds__(128)
loss_kernel(const float* __restrict__ logits, const int* __restrict__ samp,
            const float* __restrict__ proposals, const float* __restrict__ mgt,
            float* __restrict__ partials) {
  int t = blockIdx.x * blockDim.x + threadIdx.x;
  float cls = 0.0f, loc = 0.0f;
  if (t < ROWS) {
    const float* lg = logits + (size_t)t * HEADN;
    int s = samp[t];
    // softmax #1 over 81 logits
    float m1 = -3.0e38f;
    for (int i = 0; i < NCLS; ++i) m1 = fmaxf(m1, lg[i]);
    float s1 = 0.0f;
    for (int i = 0; i < NCLS; ++i) s1 += __expf(lg[i] - m1);
    float inv = 1.0f / s1;
    // log_softmax over probs: max(probs) == 1/s1 (attained at the argmax logit)
    float m2 = inv, s2 = 0.0f;
    for (int i = 0; i < NCLS; ++i) s2 += __expf(__expf(lg[i] - m1) * inv - m2);
    int tgt = (s > 0) ? s : 0;                      // clip(samp, 0)
    float pt = __expf(lg[tgt] - m1) * inv;
    float nll = -(pt - m2 - __logf(s2));
    if (s >= 0) cls = nll;
    if (s > 0) {                                    // smooth-L1 on positives only
      const float* p = proposals + (size_t)t * 4;
      const float* g = mgt + (size_t)t * 4;
      float pw = p[2] - p[0], ph = p[3] - p[1];
      float pcx = p[0] + 0.5f * pw, pcy = p[1] + 0.5f * ph;
      float gw = g[2] - g[0], gh = g[3] - g[1];
      float gcx = 0.5f * (g[0] + g[2]), gcy = 0.5f * (g[1] + g[3]);
      float gd[4] = { (gcx - pcx) / pw, (gcy - pcy) / ph,
                      __logf(gw / pw), __logf(gh / ph) };
#pragma unroll
      for (int j = 0; j < 4; ++j) {
        float d = lg[NCLS + j] - gd[j];
        float ad = fabsf(d);
        loc += (ad < 1.0f) ? 0.5f * d * d : ad - 0.5f;
      }
    }
  }
  __shared__ float rc[128], rl[128];
  rc[threadIdx.x] = cls; rl[threadIdx.x] = loc;
  __syncthreads();
  for (int o = 64; o > 0; o >>= 1) {
    if (threadIdx.x < o) { rc[threadIdx.x] += rc[threadIdx.x + o];
                           rl[threadIdx.x] += rl[threadIdx.x + o]; }
    __syncthreads();
  }
  if (threadIdx.x == 0) {
    partials[2 * blockIdx.x + 0] = rc[0];
    partials[2 * blockIdx.x + 1] = rl[0];
  }
}

__global__ void final_reduce(const float* __restrict__ partials, int nb,
                             float* __restrict__ out) {
  float c = 0.0f, l = 0.0f;
  for (int i = threadIdx.x; i < nb; i += blockDim.x) {
    c += partials[2 * i + 0]; l += partials[2 * i + 1];
  }
  __shared__ float sc[64], sl[64];
  sc[threadIdx.x] = c; sl[threadIdx.x] = l;
  __syncthreads();
  for (int o = 32; o > 0; o >>= 1) {
    if (threadIdx.x < o) { sc[threadIdx.x] += sc[threadIdx.x + o];
                           sl[threadIdx.x] += sl[threadIdx.x + o]; }
    __syncthreads();
  }
  if (threadIdx.x == 0) { out[0] = sc[0] * SCALE; out[1] = sl[0] * SCALE; }
}

// ---------------- host-side orchestration ----------------
extern "C" void kernel_launch(void* const* d_in, const int* in_sizes, int n_in,
                              void* d_out, int out_size, void* d_ws, size_t ws_size,
                              hipStream_t stream) {
  const float* X         = (const float*)d_in[0];
  const float* proposals = (const float*)d_in[1];
  const float* gt_boxes  = (const float*)d_in[2];
  const int*   gt_labels = (const int*)  d_in[3];
  const float* W1 = (const float*)d_in[4];
  const float* b1 = (const float*)d_in[5];
  const float* W2 = (const float*)d_in[6];
  const float* b2 = (const float*)d_in[7];
  const float* Wc = (const float*)d_in[8];
  const float* bc = (const float*)d_in[9];
  const float* Wb = (const float*)d_in[10];
  const float* bbv = (const float*)d_in[11];
  float* out = (float*)d_out;

  char* ws = (char*)d_ws;
  size_t off = 0;
  auto take = [&](size_t bytes) -> char* {
    char* p = ws + off;
    off = (off + bytes + 255) & ~(size_t)255;
    return p;
  };
  __bf16* Xb   = (__bf16*)take((size_t)ROWS * FD * 2);    // 200.7 MB
  __bf16* W1t  = (__bf16*)take((size_t)HID * FD * 2);     //  25.7 MB
  __bf16* W2t  = (__bf16*)take((size_t)HID * HID * 2);    //   2.1 MB
  __bf16* h1b  = (__bf16*)take((size_t)ROWS * HID * 2);   //  16.4 MB
  float*  h2f  = (float*) take((size_t)ROWS * HID * 4);   //  32.8 MB
  float*  WhT  = (float*) take((size_t)HEADN * HID * 4);
  float*  bh   = (float*) take((size_t)HEADN * 4);
  float*  logits = (float*)take((size_t)ROWS * HEADN * 4);
  int*    label = (int*)  take((size_t)ROWS * 4);
  float*  mgt   = (float*)take((size_t)ROWS * 4 * 4);
  int*    samp  = (int*)  take((size_t)ROWS * 4);
  float*  partials = (float*)take(2 * 64 * 4);

  // 1) precision conversion / weight packing
  long n4 = (long)ROWS * FD / 4;
  cvt_f32_bf16_vec4<<<(int)((n4 + 255) / 256), 256, 0, stream>>>(X, Xb, n4);
  long tw1 = (long)FD * HID;
  transpose_to_bf16<<<(int)((tw1 + 255) / 256), 256, 0, stream>>>(W1, W1t, FD, HID, tw1);
  long tw2 = (long)HID * HID;
  transpose_to_bf16<<<(int)((tw2 + 255) / 256), 256, 0, stream>>>(W2, W2t, HID, HID, tw2);
  pack_head<<<(HEADN * HID + 255) / 256, 256, 0, stream>>>(Wc, bc, Wb, bbv, WhT, bh);

  // 2) box labeling + sampling (independent of GEMMs)
  label_kernel<<<(ROWS + 127) / 128, 128, 0, stream>>>(proposals, gt_boxes, gt_labels,
                                                       label, mgt);
  sample_kernel<<<1, 64, 0, stream>>>(label, samp);

  // 3) WMMA GEMMs (bf16 in, f32 accumulate, async-LDS pipelined, fused bias+ReLU)
  dim3 gblk(256);
  dim3 ggrid(HID / 64, (ROWS + 127) / 128);
  gemm_bf16_relu<<<ggrid, gblk, 0, stream>>>(Xb, W1t, b1, nullptr, h1b, ROWS, HID, FD);
  gemm_bf16_relu<<<ggrid, gblk, 0, stream>>>(h1b, W2t, b2, h2f, nullptr, ROWS, HID, HID);

  // 4) small head + fused losses
  long th = (long)ROWS * HEADN;
  head_kernel<<<(int)((th + 127) / 128), 128, 0, stream>>>(h2f, WhT, bh, logits);
  int lb = (ROWS + 127) / 128;
  loss_kernel<<<lb, 128, 0, stream>>>(logits, samp, proposals, mgt, partials);
  final_reduce<<<1, 64, 0, stream>>>(partials, lb, out);
}
